// PairEncoder_43585328120333
// MI455X (gfx1250) — matrix operand order repaired and checked by reading, not compiled, not run on hardware
//
#include <hip/hip_runtime.h>
#include <hip/hip_bf16.h>

typedef __bf16 bf16_t;
typedef __attribute__((ext_vector_type(16))) __bf16 v16bf;
typedef __attribute__((ext_vector_type(8)))  float  v8f;

#define DMODEL 256
#define NSEQ   64
#define BATCH  2
#define NHEAD  8
#define DK     32
#define TOK    (BATCH*NSEQ*NSEQ)   /* 8192 */
#define DFF    512

__device__ __forceinline__ float bf2f(bf16_t x) {
  unsigned int v = ((unsigned int)__builtin_bit_cast(unsigned short, x)) << 16;
  return __builtin_bit_cast(float, v);
}
__device__ __forceinline__ bf16_t f2bf(float f) {
  unsigned int u = __builtin_bit_cast(unsigned int, f);
  u += 0x7fffu + ((u >> 16) & 1u);          // round-to-nearest-even
  return __builtin_bit_cast(bf16_t, (unsigned short)(u >> 16));
}

// CDNA5 async copy global -> LDS, 16B per lane, tracked by ASYNCcnt.
__device__ __forceinline__ void async_g2l_b128(unsigned lds_byte_addr, const void* gaddr) {
  asm volatile("global_load_async_to_lds_b128 %0, %1, off"
               :: "v"(lds_byte_addr), "v"(gaddr) : "memory");
}
__device__ __forceinline__ void wait_async0() {
  asm volatile("s_wait_asynccnt 0x0" ::: "memory");
}

// A fragment (16x32 bf16, ISA 7.12.2): lane row m = lane&15, half = lane>>4.
// VGPR v<4: K = 8*half + 2v + p ; v>=4: K = 16 + 8*half + 2(v-4) + p
__device__ __forceinline__ v16bf load_fragA(const bf16_t* rowp, int half) {
  v16bf f;
  const int b0 = half * 8;
#pragma unroll
  for (int v = 0; v < 4; ++v) {
    f[2*v]     = rowp[b0 + 2*v];
    f[2*v+1]   = rowp[b0 + 2*v + 1];
    f[8+2*v]   = rowp[16 + b0 + 2*v];
    f[8+2*v+1] = rowp[16 + b0 + 2*v + 1];
  }
  return f;
}
// B fragment (32x16 bf16), column-contiguous source colp[k]:
// same K order as A: VGPR v<4: K = 8*half + 2v + p ; v>=4: K = 16 + 8*half + 2(v-4) + p
__device__ __forceinline__ v16bf load_fragB(const bf16_t* colp, int half) {
  v16bf f;
  const int b0 = half * 8;
#pragma unroll
  for (int e = 0; e < 16; ++e) f[e] = colp[(e < 8 ? b0 + e : 16 + b0 + (e - 8))];
  return f;
}

__global__ void f32_to_bf16(const float* __restrict__ in, bf16_t* __restrict__ out, int n) {
  int i = blockIdx.x * blockDim.x + threadIdx.x;
  if (i < n) out[i] = f2bf(in[i]);
}

// bf16 WMMA GEMM: C[M,N] = A[M,K] @ B[K,N].
// 256 threads = 8 waves; block tile 128(M) x 64(N); wave tile 32x32 (2x2 WMMA, K-step 32).
// A tile staged via CDNA5 async global->LDS (ASYNCcnt); B tile transposed through VGPRs.
// EPI: 0 = bf16 out, 1 = f32 out, 2 = +bias,relu -> bf16, 3 = +bias -> f32
template<int EPI>
__global__ __launch_bounds__(256)
void gemm_wmma(const bf16_t* __restrict__ A, const bf16_t* __restrict__ Bm,
               const float* __restrict__ bias, void* __restrict__ Cout,
               int M, int N, int K) {
  __shared__ bf16_t As[128*32];     // row-major [m][k], 8 KB
  __shared__ bf16_t Bst[64*32];     // transposed [n][k], 4 KB
  const int tid  = threadIdx.x;
  const int lane = tid & 31;
  const int wave = tid >> 5;
  const int half = lane >> 4;
  const int mcol = lane & 15;
  const int m0 = blockIdx.y * 128;
  const int n0 = blockIdx.x * 64;
  const int mband = (wave >> 1) * 32;   // 4 M-bands
  const int nband = (wave & 1) * 32;    // 2 N-bands

  v8f c00 = {}, c01 = {}, c10 = {}, c11 = {};

  for (int k0 = 0; k0 < K; k0 += 32) {
    // ---- stage A tile 128x32 via async global->LDS (512 x 16B chunks, 2/thread)
#pragma unroll
    for (int u = 0; u < 2; ++u) {
      int t = tid + u * 256;
      int row = t >> 2, cu = t & 3;                 // 4 x 16B chunks per 32-elem row
      unsigned dst = (unsigned)(uintptr_t)&As[row*32 + cu*8];
      const bf16_t* src = A + (size_t)(m0 + row) * K + k0 + cu*8;
      async_g2l_b128(dst, (const void*)src);
    }
    // ---- stage B tile 32x64, transposed into Bst[n][k] (1024 dwords, 4/thread)
#pragma unroll
    for (int u = 0; u < 4; ++u) {
      int q = tid + u * 256;
      int kk = q >> 5, nu = q & 31;
      unsigned int val = *(const unsigned int*)(Bm + (size_t)(k0 + kk) * N + n0 + nu*2);
      Bst[(2*nu+0)*32 + kk] = __builtin_bit_cast(bf16_t, (unsigned short)(val & 0xffffu));
      Bst[(2*nu+1)*32 + kk] = __builtin_bit_cast(bf16_t, (unsigned short)(val >> 16));
    }
    wait_async0();          // ASYNCcnt -> 0: this wave's async LDS writes landed
    __syncthreads();

    v16bf a0 = load_fragA(&As[(mband      + mcol)*32], half);
    v16bf a1 = load_fragA(&As[(mband + 16 + mcol)*32], half);
    v16bf b0 = load_fragB(&Bst[(nband      + mcol)*32], half);
    v16bf b1 = load_fragB(&Bst[(nband + 16 + mcol)*32], half);
    c00 = __builtin_amdgcn_wmma_f32_16x16x32_bf16(false, a0, false, b0, (short)0, c00, false, false);
    c01 = __builtin_amdgcn_wmma_f32_16x16x32_bf16(false, a0, false, b1, (short)0, c01, false, false);
    c10 = __builtin_amdgcn_wmma_f32_16x16x32_bf16(false, a1, false, b0, (short)0, c10, false, false);
    c11 = __builtin_amdgcn_wmma_f32_16x16x32_bf16(false, a1, false, b1, (short)0, c11, false, false);
    __syncthreads();
  }

#pragma unroll
  for (int r = 0; r < 8; ++r) {
#pragma unroll
    for (int sm = 0; sm < 2; ++sm) {
      int row = m0 + mband + sm*16 + half*8 + r;
#pragma unroll
      for (int sn = 0; sn < 2; ++sn) {
        int col = n0 + nband + sn*16 + mcol;
        float v;
        if (sm == 0) v = (sn == 0) ? c00[r] : c01[r];
        else         v = (sn == 0) ? c10[r] : c11[r];
        if (EPI >= 2) v += bias[col];
        if (EPI == 2) v = v > 0.f ? v : 0.f;
        size_t idx = (size_t)row * N + col;
        if (EPI == 0 || EPI == 2) ((bf16_t*)Cout)[idx] = f2bf(v);
        else                      ((float*)Cout)[idx]  = v;
      }
    }
  }
}

// scores[b,h,i,j,l] = (1/sqrt(dk)) * sum_d q[b,i,j,h,d] * k[b,j,l,h,d]
// One block per (b,j,h): 64x64 output = 4 waves x 4 WMMA tiles, K=32 in one op.
__global__ __launch_bounds__(128)
void scores_wmma(const bf16_t* __restrict__ qb, const bf16_t* __restrict__ kb,
                 float* __restrict__ S) {
  const int idx = blockIdx.x;          // ((b*64 + j)*8 + h)
  const int h = idx & 7;
  const int j = (idx >> 3) & 63;
  const int b = idx >> 9;
  const int tid  = threadIdx.x;
  const int lane = tid & 31;
  const int wave = tid >> 5;
  const int half = lane >> 4;
  const int mcol = lane & 15;
  const int bh = b * NHEAD + h;

  const int i_row = wave * 16 + mcol;
  const bf16_t* arow = qb + (((size_t)(b*64 + i_row)*64 + j) * DMODEL) + h*DK;
  v16bf a = load_fragA(arow, half);
  const float scale = 0.17677669529663688f;   // 1/sqrt(32)
#pragma unroll
  for (int nt = 0; nt < 4; ++nt) {
    int l = nt*16 + mcol;
    const bf16_t* bcol = kb + (((size_t)(b*64 + j)*64 + l) * DMODEL) + h*DK;
    v16bf bb = load_fragB(bcol, half);
    v8f c = {};
    c = __builtin_amdgcn_wmma_f32_16x16x32_bf16(false, a, false, bb, (short)0, c, false, false);
#pragma unroll
    for (int r = 0; r < 8; ++r) {
      int ii = wave*16 + half*8 + r;
      int ll = nt*16 + mcol;
      S[(((size_t)bh*64 + ii)*64 + j)*64 + ll] = c[r] * scale;
    }
  }
}

// softmax over j for S[bh,i,j,l]; one block per (bh,i), thread per l (coalesced over l)
__global__ __launch_bounds__(64)
void softmax_j(float* __restrict__ S) {
  const int idx = blockIdx.x;           // bh*64 + i
  const int l = threadIdx.x;
  float* base = S + ((size_t)idx * 64) * 64 + l;   // j stride = 64 floats
  float mx = -1e30f;
  for (int j = 0; j < 64; ++j) mx = fmaxf(mx, base[j*64]);
  float s = 0.f;
  for (int j = 0; j < 64; ++j) { float e = __expf(base[j*64] - mx); base[j*64] = e; s += e; }
  float inv = 1.f / s;
  for (int j = 0; j < 64; ++j) base[j*64] *= inv;
}

// out[b,i,l,h,d] = sum_j att[bh,i,j,l] * v1[b,i,j,h,d] * v2[b,j,l,h,d]  (VALU, ~0.4 GFLOP)
__global__ __launch_bounds__(256)
void attn_combine(const float* __restrict__ S, const bf16_t* __restrict__ v1b,
                  const bf16_t* __restrict__ v2b, bf16_t* __restrict__ aob) {
  const int idx = blockIdx.x;           // bh*64 + i
  const int i  = idx & 63;
  const int bh = idx >> 6;
  const int h  = bh & 7;
  const int b  = bh >> 3;
  const int d  = threadIdx.x & 31;      // lane -> feature (coalesced)
  const int lq = threadIdx.x >> 5;      // wave -> l residue
  float acc[8];
#pragma unroll
  for (int t = 0; t < 8; ++t) acc[t] = 0.f;
  const float* Sbase = S + ((size_t)idx * 64) * 64;   // [j][l]
  for (int j = 0; j < 64; ++j) {
    float v1 = bf2f(v1b[(((size_t)(b*64 + i)*64 + j) * DMODEL) + h*DK + d]);
    const bf16_t* v2row = v2b + (((size_t)(b*64 + j)*64) * DMODEL) + h*DK + d;
    const float* Srow = Sbase + j*64;
#pragma unroll
    for (int t = 0; t < 8; ++t) {
      int l = lq + t*8;
      acc[t] += Srow[l] * v1 * bf2f(v2row[(size_t)l * DMODEL]);
    }
  }
#pragma unroll
  for (int t = 0; t < 8; ++t) {
    int l = lq + t*8;
    aob[(((size_t)(b*64 + i)*64 + l) * DMODEL) + h*DK + d] = f2bf(acc[t]);
  }
}

// y = LN(a + res) * g + be ; optional fp32 and bf16 outputs
__global__ __launch_bounds__(256)
void layernorm_k(const float* __restrict__ a, const float* __restrict__ res,
                 const float* __restrict__ g, const float* __restrict__ be,
                 float* __restrict__ outf, bf16_t* __restrict__ outb) {
  const int t = blockIdx.x, f = threadIdx.x;
  __shared__ float red[256];
  size_t o = (size_t)t * DMODEL + f;
  float v = a[o];
  if (res) v += res[o];
  red[f] = v; __syncthreads();
  for (int s = 128; s > 0; s >>= 1) { if (f < s) red[f] += red[f + s]; __syncthreads(); }
  float mean = red[0] * (1.f/256.f);
  __syncthreads();
  float dlt = v - mean;
  red[f] = dlt * dlt; __syncthreads();
  for (int s = 128; s > 0; s >>= 1) { if (f < s) red[f] += red[f + s]; __syncthreads(); }
  float var = red[0] * (1.f/256.f);
  float y = dlt * rsqrtf(var + 1e-5f) * g[f] + be[f];
  if (outf) outf[o] = y;
  if (outb) outb[o] = f2bf(y);
}

extern "C" void kernel_launch(void* const* d_in, const int* in_sizes, int n_in,
                              void* d_out, int out_size, void* d_ws, size_t ws_size,
                              hipStream_t stream) {
  const float* x      = (const float*)d_in[0];
  const float* Wq     = (const float*)d_in[1];
  const float* Wk     = (const float*)d_in[2];
  const float* Wv1    = (const float*)d_in[3];
  const float* Wv2    = (const float*)d_in[4];
  const float* Wo     = (const float*)d_in[5];
  const float* W1     = (const float*)d_in[6];
  const float* b1     = (const float*)d_in[7];
  const float* W2     = (const float*)d_in[8];
  const float* b2     = (const float*)d_in[9];
  const float* g_agg  = (const float*)d_in[10];
  const float* be_agg = (const float*)d_in[11];
  const float* g_out  = (const float*)d_in[12];
  const float* be_out = (const float*)d_in[13];

  char* ws = (char*)d_ws;
  size_t off = 0;
  auto alloc = [&](size_t bytes) -> char* {
    char* p = ws + off;
    off += (bytes + 255) & ~(size_t)255;
    return p;
  };
  bf16_t* xb   = (bf16_t*)alloc((size_t)TOK*DMODEL*2);
  bf16_t* wqb  = (bf16_t*)alloc((size_t)DMODEL*DMODEL*2);
  bf16_t* wkb  = (bf16_t*)alloc((size_t)DMODEL*DMODEL*2);
  bf16_t* wv1b = (bf16_t*)alloc((size_t)DMODEL*DMODEL*2);
  bf16_t* wv2b = (bf16_t*)alloc((size_t)DMODEL*DMODEL*2);
  bf16_t* wob  = (bf16_t*)alloc((size_t)DMODEL*DMODEL*2);
  bf16_t* w1b  = (bf16_t*)alloc((size_t)DMODEL*DFF*2);
  bf16_t* w2b  = (bf16_t*)alloc((size_t)DFF*DMODEL*2);
  bf16_t* qb   = (bf16_t*)alloc((size_t)TOK*DMODEL*2);
  bf16_t* kb   = (bf16_t*)alloc((size_t)TOK*DMODEL*2);
  bf16_t* v1b  = (bf16_t*)alloc((size_t)TOK*DMODEL*2);
  bf16_t* v2b  = (bf16_t*)alloc((size_t)TOK*DMODEL*2);
  float*  S    = (float*) alloc((size_t)BATCH*NHEAD*NSEQ*NSEQ*NSEQ*4); // 16 MB
  bf16_t* aob  = (bf16_t*)alloc((size_t)TOK*DMODEL*2);
  float*  ar   = (float*) alloc((size_t)TOK*DMODEL*4);
  float*  hbuf = (float*) alloc((size_t)TOK*DMODEL*4);
  bf16_t* hb   = (bf16_t*)alloc((size_t)TOK*DMODEL*2);
  bf16_t* ffb  = (bf16_t*)alloc((size_t)TOK*DFF*2);
  float*  y2   = (float*) alloc((size_t)TOK*DMODEL*4);
  if (off > ws_size) return;   // workspace too small; bail without OOB writes

  auto cvt = [&](const float* src, bf16_t* dst, int n) {
    f32_to_bf16<<<(n + 255) / 256, 256, 0, stream>>>(src, dst, n);
  };
  cvt(x,   xb,   TOK*DMODEL);
  cvt(Wq,  wqb,  DMODEL*DMODEL);
  cvt(Wk,  wkb,  DMODEL*DMODEL);
  cvt(Wv1, wv1b, DMODEL*DMODEL);
  cvt(Wv2, wv2b, DMODEL*DMODEL);
  cvt(Wo,  wob,  DMODEL*DMODEL);
  cvt(W1,  w1b,  DMODEL*DFF);
  cvt(W2,  w2b,  DFF*DMODEL);

  // projections: [8192,256] @ [256,256] -> bf16
  dim3 gP(DMODEL/64, TOK/128);
  gemm_wmma<0><<<gP, 256, 0, stream>>>(xb, wqb,  nullptr, qb,  TOK, DMODEL, DMODEL);
  gemm_wmma<0><<<gP, 256, 0, stream>>>(xb, wkb,  nullptr, kb,  TOK, DMODEL, DMODEL);
  gemm_wmma<0><<<gP, 256, 0, stream>>>(xb, wv1b, nullptr, v1b, TOK, DMODEL, DMODEL);
  gemm_wmma<0><<<gP, 256, 0, stream>>>(xb, wv2b, nullptr, v2b, TOK, DMODEL, DMODEL);

  // attention
  scores_wmma<<<BATCH*NSEQ*NHEAD, 128, 0, stream>>>(qb, kb, S);
  softmax_j<<<BATCH*NHEAD*NSEQ, 64, 0, stream>>>(S);
  attn_combine<<<BATCH*NHEAD*NSEQ, 256, 0, stream>>>(S, v1b, v2b, aob);

  // output projection + LN1
  gemm_wmma<1><<<gP, 256, 0, stream>>>(aob, wob, nullptr, ar, TOK, DMODEL, DMODEL);
  layernorm_k<<<TOK, 256, 0, stream>>>(ar, x, g_agg, be_agg, hbuf, hb);

  // FFN + LN2
  dim3 g1(DFF/64, TOK/128);
  gemm_wmma<2><<<g1, 256, 0, stream>>>(hb, w1b, b1, ffb, TOK, DFF, DMODEL);
  dim3 g2(DMODEL/64, TOK/128);
  gemm_wmma<3><<<g2, 256, 0, stream>>>(ffb, w2b, b2, y2, TOK, DMODEL, DFF);
  layernorm_k<<<TOK, 256, 0, stream>>>(y2, hbuf, g_out, be_out, (float*)d_out, (bf16_t*)nullptr);
}
